// System2Graph_60687887892696
// MI455X (gfx1250) — matrix operand order repaired and checked by reading, not compiled, not run on hardware
//
#include <hip/hip_runtime.h>
#include <math.h>

#define Bq 8
#define Tq 4096
#define Dq 512
#define TWO_D 1024
#define NWG 16
#define NTHREADS 256

typedef __attribute__((ext_vector_type(16))) __bf16 v16bf;
typedef __attribute__((ext_vector_type(8)))  __bf16 v8bf;
typedef __attribute__((ext_vector_type(8)))  float  v8f;
typedef __attribute__((ext_vector_type(4)))  unsigned int v4u;
typedef __attribute__((ext_vector_type(8)))  int    v8i;
typedef __attribute__((ext_vector_type(4)))  int    v4i;

__device__ __forceinline__ unsigned short f2bf(float f) {
  unsigned u = __builtin_bit_cast(unsigned int, f);
  unsigned r = 0x7FFFu + ((u >> 16) & 1u);
  return (unsigned short)((u + r) >> 16);
}

// A-operand fragment (16x32 bf16, MxK): lane half hf=0 holds K[kb..kb+15];
// hf=1 holds K[kb+16..kb+31] (contiguous, matching B fragment).
__device__ __forceinline__ v16bf load_a_frag(const unsigned short* row, int kb, int hf) {
  v8bf lo = *(const v8bf*)(row + kb + hf * 16);
  v8bf hi = *(const v8bf*)(row + kb + hf * 16 + 8);
  v16bf o;
#pragma unroll
  for (int i = 0; i < 8; ++i) { o[i] = lo[i]; o[i + 8] = hi[i]; }
  return o;
}

// B-operand fragment (32x16 bf16, KxN): lane = N (+16 for upper K half);
// per-lane 16 consecutive K values -> single 32B chunk of W[n][kb + 16*hf ...].
__device__ __forceinline__ v16bf load_b_frag(const unsigned short* row, int kb, int hf) {
  return *(const v16bf*)(row + kb + hf * 16);
}

// Tensor Data Mover: 2-D tile load global -> LDS (bf16/2-byte elements).
// D# built per ISA cdna5 ch.8: group0 = {count|flags, lds_addr, gaddr_lo,
// gaddr_hi|type=2}; group1 = {data_size, tensor_dim0/1, tile_dim0/1, stride0}.
// This toolchain's builtin takes 6 args: (v4u, v8i, v4i, v4i, v8i, i32 cpol).
__device__ __forceinline__ void tdm_load_2d(unsigned lds_off, const void* gptr,
                                            unsigned width, unsigned height,
                                            unsigned stride) {
  unsigned long long ga = (unsigned long long)gptr;
  v4u g0;
  g0[0] = 1u;                                              // count=1 (valid), user mode
  g0[1] = lds_off;                                         // lds_addr (bytes)
  g0[2] = (unsigned)(ga & 0xFFFFFFFFu);                    // global_addr[31:0]
  g0[3] = (unsigned)((ga >> 32) & 0x1FFFFFFu) | 0x80000000u; // addr[56:32] | type=2
  v8i g1;
  g1[0] = (int)(1u << 16);                                 // data_size=1 (2B), mask=0
  g1[1] = (int)((width & 0xFFFFu) << 16);                  // tensor_dim0[15:0]
  g1[2] = (int)(((width >> 16) & 0xFFFFu) | ((height & 0xFFFFu) << 16)); // dim0 hi|dim1 lo
  g1[3] = (int)(((height >> 16) & 0xFFFFu) | ((width & 0xFFFFu) << 16)); // dim1 hi|tile_dim0
  g1[4] = (int)(height & 0xFFFFu);                         // tile_dim1 (tile_dim2=0)
  g1[5] = (int)stride;                                     // tensor_dim0_stride[31:0]
  g1[6] = 0;                                               // stride0 hi | stride1 lo
  g1[7] = 0;
  v4i z4 = {0, 0, 0, 0};
  v8i z8 = {0, 0, 0, 0, 0, 0, 0, 0};
  __builtin_amdgcn_tensor_load_to_lds(g0, g1, z4, z4, z8, 0);
}

__device__ __forceinline__ void device_barrier(unsigned* bar) {
  __threadfence();                       // flush this wave's stores to device scope
  __builtin_amdgcn_s_cluster_barrier();  // s_barrier_signal/-wait -3 (NOP if not clustered)
  __syncthreads();
  if (threadIdx.x == 0) {
    unsigned* cnt = bar;
    unsigned* gen = bar + 1;
    unsigned g = __hip_atomic_load(gen, __ATOMIC_ACQUIRE, __HIP_MEMORY_SCOPE_AGENT);
    unsigned a = __hip_atomic_fetch_add(cnt, 1u, __ATOMIC_ACQ_REL, __HIP_MEMORY_SCOPE_AGENT);
    if (a == NWG - 1) {
      __hip_atomic_store(cnt, 0u, __ATOMIC_RELAXED, __HIP_MEMORY_SCOPE_AGENT);
      __hip_atomic_fetch_add(gen, 1u, __ATOMIC_RELEASE, __HIP_MEMORY_SCOPE_AGENT);
    } else {
      while (__hip_atomic_load(gen, __ATOMIC_ACQUIRE, __HIP_MEMORY_SCOPE_AGENT) == g) {
        __builtin_amdgcn_s_sleep(2);
      }
    }
  }
  __syncthreads();
  __threadfence();                       // acquire: invalidate stale near-cache lines
}

__global__ void prep_kernel(const float* __restrict__ W1, const float* __restrict__ W2,
                            unsigned short* __restrict__ w1b, unsigned short* __restrict__ w2b,
                            float* __restrict__ g_mem, unsigned* __restrict__ bar) {
  int i = blockIdx.x * blockDim.x + threadIdx.x;
  int stride = gridDim.x * blockDim.x;
  for (int k = i; k < TWO_D * TWO_D; k += stride) w1b[k] = f2bf(W1[k]);
  for (int k = i; k < Dq * TWO_D; k += stride) w2b[k] = f2bf(W2[k]);
  for (int k = i; k < Bq * Dq; k += stride) g_mem[k] = 0.0f;
  if (i < 2) bar[i] = 0u;
}

__global__ __launch_bounds__(NTHREADS)
void scan_kernel(const float* __restrict__ x,
                 const unsigned short* __restrict__ w1b,
                 const unsigned short* __restrict__ w2b,
                 const float* __restrict__ b1,
                 const float* __restrict__ b2,
                 unsigned short* __restrict__ g_h,
                 float* __restrict__ g_mem,
                 unsigned* __restrict__ bar,
                 float* __restrict__ out) {
  // dynamic LDS: edge[16][1024] + h[16][1024] + W1slice[64][1024] + W2slice[32][1024] bf16
  extern __shared__ unsigned short dynsm[];
  unsigned short* lds_edge = dynsm;                 // 32 KB
  unsigned short* lds_h    = dynsm + 16 * 1024;     // 32 KB
  unsigned short* lds_w1   = dynsm + 32 * 1024;     // 128 KB
  unsigned short* lds_w2   = dynsm + 96 * 1024;     // 64 KB
  __shared__ float lds_acc[6 * 256];                // 4 GEMM1 tiles + 2 GEMM2 tiles

  const int tid  = threadIdx.x;
  const int wg   = blockIdx.x;
  const int wave = tid >> 5;
  const int lane = tid & 31;
  const int hf   = lane >> 4;
  const int m16  = lane & 15;

  // ---- one-time: stage resident weight slices into LDS via the Tensor Data Mover ----
  if (wave == 0) {
    tdm_load_2d((unsigned)(uintptr_t)(void*)lds_w1, w1b + (size_t)wg * 64 * 1024,
                1024u, 64u, 1024u);
    tdm_load_2d((unsigned)(uintptr_t)(void*)lds_w2, w2b + (size_t)wg * 32 * 1024,
                1024u, 32u, 1024u);
    __builtin_amdgcn_s_wait_tensorcnt(0);
  }
  // zero padding rows (M=8..15) of both A operands
  for (int i = tid; i < 8 * 1024; i += NTHREADS) {
    lds_edge[8 * 1024 + i] = 0;
    lds_h[8 * 1024 + i] = 0;
  }
  // initial edge: x[:,0,:] in cols [0,512), mem=0 in cols [512,1024)
  {
    int b = tid >> 5;
    int c0 = (tid & 31) * 16;
    const float* xr = x + (size_t)b * Tq * Dq + c0;
#pragma unroll
    for (int j = 0; j < 16; ++j) {
      lds_edge[b * 1024 + c0 + j] = f2bf(xr[j]);
      lds_edge[b * 1024 + 512 + c0 + j] = 0;
    }
  }
  // per-thread bias registers (constant over t)
  float b1v[4], b2v[2];
  {
    int nc = tid & 15;
#pragma unroll
    for (int tl = 0; tl < 4; ++tl) b1v[tl] = b1[wg * 64 + tl * 16 + nc];
#pragma unroll
    for (int tl = 0; tl < 2; ++tl) b2v[tl] = b2[wg * 32 + tl * 16 + nc];
  }
  __syncthreads();

  for (int t = 0; t < Tq; ++t) {
    for (int i = tid; i < 6 * 256; i += NTHREADS) lds_acc[i] = 0.0f;
    __syncthreads();

    // ---- GEMM1: edge[16x1024] @ W1slice[64x1024]^T ; 8 waves = 4 N-tiles x 2 K-halves ----
    {
      int tile = wave & 3;
      int kh = wave >> 2;
      const unsigned short* arow = lds_edge + m16 * 1024;
      const unsigned short* brow = lds_w1 + (tile * 16 + m16) * 1024;
      v8f acc = {0.f, 0.f, 0.f, 0.f, 0.f, 0.f, 0.f, 0.f};
      for (int kb = kh * 512; kb < kh * 512 + 512; kb += 32) {
        v16bf a = load_a_frag(arow, kb, hf);
        v16bf bm = load_b_frag(brow, kb, hf);
        acc = __builtin_amdgcn_wmma_f32_16x16x32_bf16(false, a, false, bm, (short)0, acc,
                                                      false, false);
      }
      float* accb = lds_acc + tile * 256;
#pragma unroll
      for (int r = 0; r < 8; ++r) atomicAdd(&accb[(r + 8 * hf) * 16 + m16], acc[r]);
    }
    __syncthreads();

    // ---- epilogue 1: bias + exact GELU -> g_h slice (bf16) ----
    {
      int mr = tid >> 4, nc = tid & 15;
#pragma unroll
      for (int tl = 0; tl < 4; ++tl) {
        if (mr < 8) {
          float v = lds_acc[tl * 256 + tid] + b1v[tl];
          float h = 0.5f * v * (1.0f + erff(v * 0.70710678118654752f));
          g_h[mr * 1024 + wg * 64 + tl * 16 + nc] = f2bf(h);
        }
      }
    }
    device_barrier(bar);

    // gather full h [8x1024] from all workgroups into LDS via TDM async DMA
    if (wave == 0) {
      tdm_load_2d((unsigned)(uintptr_t)(void*)lds_h, g_h, 1024u, 8u, 1024u);
      __builtin_amdgcn_s_wait_tensorcnt(0);
    }
    __syncthreads();

    // ---- GEMM2: h[16x1024] @ W2slice[32x1024]^T ; waves 0-3 = 2 N-tiles x 2 K-halves ----
    if (wave < 4) {
      int tile = wave & 1;
      int kh = wave >> 1;
      const unsigned short* arow = lds_h + m16 * 1024;
      const unsigned short* brow = lds_w2 + (tile * 16 + m16) * 1024;
      v8f acc = {0.f, 0.f, 0.f, 0.f, 0.f, 0.f, 0.f, 0.f};
      for (int kb = kh * 512; kb < kh * 512 + 512; kb += 32) {
        v16bf a = load_a_frag(arow, kb, hf);
        v16bf bm = load_b_frag(brow, kb, hf);
        acc = __builtin_amdgcn_wmma_f32_16x16x32_bf16(false, a, false, bm, (short)0, acc,
                                                      false, false);
      }
      float* accb = lds_acc + 1024 + tile * 256;
#pragma unroll
      for (int r = 0; r < 8; ++r) atomicAdd(&accb[(r + 8 * hf) * 16 + m16], acc[r]);
    }
    __syncthreads();

    // ---- epilogue 2: bias + sigmoid gate + mem update -> g_mem slice, out[b,t,:] ----
    {
      int mr = tid >> 4, nc = tid & 15;
#pragma unroll
      for (int tl = 0; tl < 2; ++tl) {
        if (mr < 8) {
          float prop = lds_acc[1024 + tl * 256 + tid] + b2v[tl];
          float g = 1.0f / (1.0f + expf(-prop));
          int mg = wg * 32 + tl * 16 + nc;
          float old = g_mem[mr * 512 + mg];
          float nm = old + g * (prop - old);
          g_mem[mr * 512 + mg] = nm;
          out[(size_t)mr * Tq * Dq + (size_t)t * Dq + mg] = nm;
        }
      }
    }
    device_barrier(bar);

    // refresh edge A-operand: new mem (all 512 cols) + next x[:,t+1,:]
    {
      int b = tid >> 5;
      int c0 = (tid & 31) * 16;
      const float* mp = g_mem + b * 512 + c0;
#pragma unroll
      for (int j = 0; j < 16; ++j) lds_edge[b * 1024 + 512 + c0 + j] = f2bf(mp[j]);
      if (t + 1 < Tq) {
        const float* xr = x + (size_t)b * Tq * Dq + (size_t)(t + 1) * Dq + c0;
#pragma unroll
        for (int j = 0; j < 16; ++j) lds_edge[b * 1024 + c0 + j] = f2bf(xr[j]);
        if (t + 2 < Tq)
          __builtin_prefetch(x + (size_t)b * Tq * Dq + (size_t)(t + 2) * Dq + c0, 0, 1);
      }
    }
    __syncthreads();
  }
}

extern "C" void kernel_launch(void* const* d_in, const int* in_sizes, int n_in,
                              void* d_out, int out_size, void* d_ws, size_t ws_size,
                              hipStream_t stream) {
  const float* x  = (const float*)d_in[0];
  const float* W1 = (const float*)d_in[1];
  const float* b1 = (const float*)d_in[2];
  const float* W2 = (const float*)d_in[3];
  const float* b2 = (const float*)d_in[4];
  float* out = (float*)d_out;

  char* ws = (char*)d_ws;
  unsigned short* w1b = (unsigned short*)ws;                              // 2 MB bf16 W1
  unsigned short* w2b = (unsigned short*)(ws + (2u << 20));               // 1 MB bf16 W2
  unsigned short* g_h = (unsigned short*)(ws + (3u << 20));               // 16 KB h exchange
  float* g_mem = (float*)(ws + (3u << 20) + (16u << 10));                 // 16 KB mem state
  unsigned* bar = (unsigned*)(ws + (3u << 20) + (32u << 10));             // barrier cnt/gen

  prep_kernel<<<512, 256, 0, stream>>>(W1, W2, w1b, w2b, g_mem, bar);

  // 256 KB dynamic LDS per workgroup (CDNA5 WGP has 320 KB)
  (void)hipFuncSetAttribute(reinterpret_cast<const void*>(scan_kernel),
                            hipFuncAttributeMaxDynamicSharedMemorySize, 262144);
  scan_kernel<<<NWG, NTHREADS, 262144, stream>>>(x, w1b, w2b, b1, b2, g_h, g_mem, bar, out);
}
